// TemporalHGT_19885698581032
// MI455X (gfx1250) — compile-verified, matrix operand order checked by reading, not run on hardware
//
#include <hip/hip_runtime.h>

// ---------------- problem constants (match reference) ----------------
#define N_NODES 100000
#define F_IN    64
#define H_DIM   128
#define NHEAD   4
#define DHEAD   32
#define NTYPE   5
#define ETYPE   10
#define NGRP    50
#define NLAYERS 3
#define OUT_DIM 8
#define LMAXV   700
#define LTILES  11   // ceil(700/64)
#define MTILES  22   // ceil(700/32)

#define INW_CNT  (NLAYERS * ETYPE * 3 * H_DIM * H_DIM)  // 1,474,560
#define OUTW_CNT (NLAYERS * ETYPE * H_DIM * H_DIM)      //   491,520

typedef __attribute__((ext_vector_type(16))) __bf16 v16bf;
typedef __attribute__((ext_vector_type(8)))  float  v8f;

union Frag16 { v16bf v; unsigned u[8]; };

// hardware bf16 conversion (v_cvt_*bf16_f32), not software RNE
__device__ __forceinline__ unsigned short f2bf(float x) {
  return __builtin_bit_cast(unsigned short, (__bf16)x);
}
// K index of dword j for the 16-bit A/B fragment layout (ISA 7.12.2)
__device__ __forceinline__ int kmap(int j, int half) {
  return (j < 4) ? (2 * j + 8 * half) : (16 + 2 * (j - 4) + 8 * half);
}
// fragment from a contiguous bf16 row (row = this lane's M or N index) - pure loads
__device__ __forceinline__ v16bf frag_from_bf16(const unsigned short* row, int half) {
  Frag16 f;
#pragma unroll
  for (int j = 0; j < 8; ++j) f.u[j] = *(const unsigned*)(row + kmap(j, half));
  return f.v;
}
__device__ __forceinline__ v8f wmma_bf16(v16bf a, v16bf b, v8f c) {
  return __builtin_amdgcn_wmma_f32_16x16x32_bf16(false, a, false, b, (short)0, c, false, false);
}
// CDNA5 async global->LDS copy (ASYNCcnt); 16B per lane
__device__ __forceinline__ void async_copy_b128(unsigned lds_addr, const void* gptr) {
  asm volatile("global_load_async_to_lds_b128 %0, %1, off"
               :: "v"(lds_addr), "v"((unsigned long long)(uintptr_t)gptr) : "memory");
}
__device__ __forceinline__ void wait_async() {
  asm volatile("s_wait_asynccnt 0x0" ::: "memory");
}

// ---------------- one-time prep: bf16 copies of all GEMM weights ----------------
__global__ void prep_weights(const float* __restrict__ in_w, const float* __restrict__ out_w,
                             const float* __restrict__ w1, const float* __restrict__ w2,
                             unsigned short* __restrict__ in_w_bf,
                             unsigned short* __restrict__ out_w_bf,
                             unsigned short* __restrict__ w1T, unsigned short* __restrict__ w2T) {
  int stride = gridDim.x * blockDim.x;
  int tid = blockIdx.x * blockDim.x + threadIdx.x;
  for (int i = tid; i < INW_CNT; i += stride) in_w_bf[i] = f2bf(in_w[i]);
  for (int i = tid; i < OUTW_CNT; i += stride) out_w_bf[i] = f2bf(out_w[i]);
  for (int i = tid; i < H_DIM * H_DIM; i += stride) {      // w1T[e][d] = w1[d][e]
    int e = i / H_DIM, d = i % H_DIM;
    w1T[e * H_DIM + d] = f2bf(w1[d * H_DIM + e]);
  }
  for (int i = tid; i < 16 * H_DIM; i += stride) {         // w2T[o][e] = w2[e][o], pad to 16
    int o = i / H_DIM, e = i % H_DIM;
    w2T[o * H_DIM + e] = (o < OUT_DIM) ? f2bf(w2[e * OUT_DIM + o]) : (unsigned short)0;
  }
}

// ---------------- per-node-type input projection ----------------
__global__ void input_proj(const float* __restrict__ x, const long long* __restrict__ node_type,
                           const float* __restrict__ inp_w, const float* __restrict__ inp_b,
                           float* __restrict__ h, unsigned short* __restrict__ h_bf) {
  int wv = threadIdx.x >> 5, lane = threadIdx.x & 31;
  int n = blockIdx.x * 8 + wv;
  if (n >= N_NODES) return;
  int t = (int)node_type[n];
  const float* W  = inp_w + (size_t)t * F_IN * H_DIM;
  const float* xr = x + (size_t)n * F_IN;
  float acc[4];
#pragma unroll
  for (int k = 0; k < 4; ++k) acc[k] = inp_b[t * H_DIM + lane + 32 * k];
  for (int f = 0; f < F_IN; ++f) {
    float xv = xr[f];
#pragma unroll
    for (int k = 0; k < 4; ++k) acc[k] += xv * W[f * H_DIM + lane + 32 * k];
  }
#pragma unroll
  for (int k = 0; k < 4; ++k) {
    size_t oi = (size_t)n * H_DIM + lane + 32 * k;
    h[oi] = acc[k];
    h_bf[oi] = f2bf(acc[k]);
  }
}

// ---------------- QKV: gathered bf16 rows DMA'd into LDS (async), then WMMA ----------------
__global__ __launch_bounds__(256) void qkv_kernel(
    const unsigned short* __restrict__ h_bf, const int* __restrict__ group_src,
    const int* __restrict__ group_dst, const unsigned short* __restrict__ in_w_bf,
    const float* __restrict__ in_b, int layer,
    unsigned short* __restrict__ q_bf, unsigned short* __restrict__ k_bf,
    unsigned short* __restrict__ vT_bf) {
  __shared__ __attribute__((aligned(16))) unsigned short a_dst[64 * H_DIM]; // 16KB
  __shared__ __attribute__((aligned(16))) unsigned short a_src[64 * H_DIM]; // 16KB
  int g = blockIdx.x / LTILES, lt = blockIdx.x % LTILES;
  int et = g / NTYPE;
  int l0 = lt * 64;
  { // async gather: each thread copies 4x16B per matrix for its (row, quarter)
    int r = threadIdx.x >> 2, q4 = threadIdx.x & 3;
    int l = l0 + r;
    int is = (l < LMAXV) ? group_src[g * LMAXV + l] : 0;
    int id = (l < LMAXV) ? group_dst[g * LMAXV + l] : 0;
    const unsigned short* gs = h_bf + (size_t)is * H_DIM + q4 * 32; // 64B span
    const unsigned short* gd = h_bf + (size_t)id * H_DIM + q4 * 32;
    unsigned ls = (unsigned)(uintptr_t)&a_src[r * H_DIM + q4 * 32];
    unsigned ld = (unsigned)(uintptr_t)&a_dst[r * H_DIM + q4 * 32];
#pragma unroll
    for (int c = 0; c < 4; ++c) {
      async_copy_b128(ls + c * 16, gs + c * 8);
      async_copy_b128(ld + c * 16, gd + c * 8);
    }
    wait_async();
  }
  __syncthreads();
  int wv = threadIdx.x >> 5, lane = threadIdx.x & 31;
  int nIdx = lane & 15, half = lane >> 4;
  int eCol = wv * 16 + nIdx;
  const unsigned short* Wbase = in_w_bf + (((size_t)layer * ETYPE + et) * 3 * H_DIM) * H_DIM;
  const float* Bbase = in_b + ((size_t)layer * ETYPE + et) * 3 * H_DIM;
  for (int p = 0; p < 3; ++p) {          // 0=q(dst), 1=k(src), 2=v(src)
    const unsigned short* Wrow = Wbase + ((size_t)p * H_DIM + eCol) * H_DIM; // W[e][d]
    v16bf bfr[4];
#pragma unroll
    for (int ks = 0; ks < 4; ++ks) bfr[ks] = frag_from_bf16(Wrow + ks * 32, half);
    float bias = Bbase[p * H_DIM + eCol];
    const unsigned short* A = (p == 0) ? a_dst : a_src;
#pragma unroll
    for (int mt = 0; mt < 4; ++mt) {
      v8f c = {};
#pragma unroll
      for (int ks = 0; ks < 4; ++ks) {
        v16bf a = frag_from_bf16(A + (mt * 16 + nIdx) * H_DIM + ks * 32, half);
        c = wmma_bf16(a, bfr[ks], c);
      }
#pragma unroll
      for (int r = 0; r < 8; ++r) {
        int l = l0 + mt * 16 + r + 8 * half;
        if (l < LMAXV) {
          unsigned short val = f2bf(c[r] + bias);
          size_t gi = (size_t)g;
          if (p == 0)      q_bf[(gi * LMAXV + l) * H_DIM + eCol] = val;
          else if (p == 1) k_bf[(gi * LMAXV + l) * H_DIM + eCol] = val;
          else             vT_bf[(gi * H_DIM + eCol) * LMAXV + l] = val; // transposed for PV
        }
      }
    }
  }
}

// ---------------- attention: flash-style, wave = (16 q-rows, 1 head) ----------------
__global__ __launch_bounds__(256) void attn_kernel(
    const unsigned short* __restrict__ q_bf, const unsigned short* __restrict__ k_bf,
    const unsigned short* __restrict__ vT_bf, const unsigned char* __restrict__ group_valid,
    unsigned short* __restrict__ o_bf) {
  __shared__ __attribute__((aligned(16))) unsigned short ptile[8][16 * 32]; // bf16 P, 8KB
  int g = blockIdx.x / MTILES, lt = blockIdx.x % MTILES;
  int wv = threadIdx.x >> 5, lane = threadIdx.x & 31;
  int nIdx = lane & 15, half = lane >> 4;
  int head = wv & 3, lsub = wv >> 2;
  int l0 = lt * 32 + lsub * 16;
  int lq = l0 + nIdx; if (lq >= LMAXV) lq = 0;
  v16bf aq = frag_from_bf16(q_bf + ((size_t)g * LMAXV + lq) * H_DIM + head * DHEAD, half);
  float m_run[8], l_run[8];
  v8f o0 = {}, o1 = {};
#pragma unroll
  for (int r = 0; r < 8; ++r) { m_run[r] = -1e30f; l_run[r] = 0.f; }
  const float scale = 0.17677669529663687f;      // 1/sqrt(32)
  unsigned short* pt = ptile[wv];
  for (int ms = 0; ms < MTILES; ++ms) {
    int mb = ms * 32;
    int m0 = mb + nIdx, m1 = mb + 16 + nIdx;
    int m0c = (m0 < LMAXV) ? m0 : 0, m1c = (m1 < LMAXV) ? m1 : 0;
    v16bf bk0 = frag_from_bf16(k_bf + ((size_t)g * LMAXV + m0c) * H_DIM + head * DHEAD, half);
    v16bf bk1 = frag_from_bf16(k_bf + ((size_t)g * LMAXV + m1c) * H_DIM + head * DHEAD, half);
    // prefetch next K tile rows (global_prefetch_b8)
    if (m0 + 32 < LMAXV)
      __builtin_prefetch(k_bf + ((size_t)g * LMAXV + m0 + 32) * H_DIM + head * DHEAD, 0, 1);
    v8f z = {};
    v8f s0 = wmma_bf16(aq, bk0, z);              // D: rows=l, cols(lanes)=m
    v8f s1 = wmma_bf16(aq, bk1, z);
    bool v0 = (m0 < LMAXV) && group_valid[g * LMAXV + m0c];
    bool v1 = (m1 < LMAXV) && group_valid[g * LMAXV + m1c];
    float corr[8];
#pragma unroll
    for (int r = 0; r < 8; ++r) {
      float a0 = v0 ? s0[r] * scale : -1e9f;
      float a1 = v1 ? s1[r] * scale : -1e9f;
      float mx = fmaxf(a0, a1);                  // row-max across the 16-lane half
      mx = fmaxf(mx, __shfl_xor(mx, 1, 32));
      mx = fmaxf(mx, __shfl_xor(mx, 2, 32));
      mx = fmaxf(mx, __shfl_xor(mx, 4, 32));
      mx = fmaxf(mx, __shfl_xor(mx, 8, 32));
      float nm = fmaxf(m_run[r], mx);
      corr[r] = __expf(m_run[r] - nm);
      m_run[r] = nm;
      float p0 = __expf(a0 - nm), p1 = __expf(a1 - nm);
      float ps = p0 + p1;
      ps += __shfl_xor(ps, 1, 32);
      ps += __shfl_xor(ps, 2, 32);
      ps += __shfl_xor(ps, 4, 32);
      ps += __shfl_xor(ps, 8, 32);
      l_run[r] = l_run[r] * corr[r] + ps;
      int lrow = r + 8 * half;
      pt[lrow * 32 + nIdx]      = f2bf(p0);      // bf16 P tile in LDS
      pt[lrow * 32 + 16 + nIdx] = f2bf(p1);
    }
#pragma unroll
    for (int r = 0; r < 8; ++r) { o0[r] *= corr[r]; o1[r] *= corr[r]; }
    // P as A-fragment (rows=l, K=32 m's) from wave-private LDS (in-order within wave)
    v16bf ap = frag_from_bf16(pt + nIdx * 32, half);
    const unsigned short* vbase = vT_bf + ((size_t)g * H_DIM + head * DHEAD) * LMAXV;
    v16bf bv0 = frag_from_bf16(vbase + (size_t)nIdx * LMAXV + mb, half);
    v16bf bv1 = frag_from_bf16(vbase + (size_t)(nIdx + 16) * LMAXV + mb, half);
    if (mb + 32 < LMAXV)
      __builtin_prefetch(vbase + (size_t)nIdx * LMAXV + mb + 32, 0, 1);
    o0 = wmma_bf16(ap, bv0, o0);
    o1 = wmma_bf16(ap, bv1, o1);
  }
#pragma unroll
  for (int r = 0; r < 8; ++r) {
    float inv = (l_run[r] > 0.f) ? 1.f / l_run[r] : 0.f;
    int l = l0 + r + 8 * half;
    if (l < LMAXV) {
      size_t base = ((size_t)g * LMAXV + l) * H_DIM + head * DHEAD;
      o_bf[base + nIdx]      = f2bf(o0[r] * inv);
      o_bf[base + 16 + nIdx] = f2bf(o1[r] * inv);
    }
  }
}

// ---------------- output projection + masked scatter-add (f32 atomics) ----------------
__global__ __launch_bounds__(256) void outproj_kernel(
    const unsigned short* __restrict__ o_bf, const int* __restrict__ group_dst,
    const unsigned char* __restrict__ group_valid, const unsigned short* __restrict__ out_w_bf,
    const float* __restrict__ out_b, int layer, float* __restrict__ h_acc) {
  int g = blockIdx.x / LTILES, lt = blockIdx.x % LTILES;
  int et = g / NTYPE;
  int wv = threadIdx.x >> 5, lane = threadIdx.x & 31;
  int nIdx = lane & 15, half = lane >> 4;
  int fCol = wv * 16 + nIdx;
  const unsigned short* Wrow =
      out_w_bf + ((((size_t)layer * ETYPE + et) * H_DIM) + fCol) * H_DIM;  // Wo[f][e]
  v16bf bfr[4];
#pragma unroll
  for (int ks = 0; ks < 4; ++ks) bfr[ks] = frag_from_bf16(Wrow + ks * 32, half);
  float bias = out_b[((size_t)layer * ETYPE + et) * H_DIM + fCol];
  int l0 = lt * 64;
  for (int mt = 0; mt < 4; ++mt) {
    int lrow = l0 + mt * 16 + nIdx; if (lrow >= LMAXV) lrow = 0;
    const unsigned short* Arow = o_bf + ((size_t)g * LMAXV + lrow) * H_DIM;
    v8f c = {};
#pragma unroll
    for (int ks = 0; ks < 4; ++ks)
      c = wmma_bf16(frag_from_bf16(Arow + ks * 32, half), bfr[ks], c);
#pragma unroll
    for (int r = 0; r < 8; ++r) {
      int l = l0 + mt * 16 + r + 8 * half;
      if (l < LMAXV && group_valid[g * LMAXV + l]) {
        int dst = group_dst[g * LMAXV + l];
        atomicAdd(&h_acc[(size_t)dst * H_DIM + fCol], c[r] + bias);
      }
    }
  }
}

// ---------------- layernorm + relu + residual (wave per node); keeps bf16 mirror ----------------
__global__ void ln_update(const float* __restrict__ h_acc, const float* __restrict__ ln_g,
                          const float* __restrict__ ln_b, int layer, float* __restrict__ h,
                          unsigned short* __restrict__ h_bf) {
  int wv = threadIdx.x >> 5, lane = threadIdx.x & 31;
  int n = blockIdx.x * 8 + wv;
  if (n >= N_NODES) return;
  float v[4], s = 0.f;
#pragma unroll
  for (int k = 0; k < 4; ++k) { v[k] = h_acc[(size_t)n * H_DIM + lane + 32 * k]; s += v[k]; }
#pragma unroll
  for (int m = 1; m < 32; m <<= 1) s += __shfl_xor(s, m, 32);
  float mu = s * (1.f / H_DIM), vs = 0.f;
#pragma unroll
  for (int k = 0; k < 4; ++k) { float d = v[k] - mu; vs += d * d; }
#pragma unroll
  for (int m = 1; m < 32; m <<= 1) vs += __shfl_xor(vs, m, 32);
  float inv = rsqrtf(vs * (1.f / H_DIM) + 1e-5f);
#pragma unroll
  for (int k = 0; k < 4; ++k) {
    int c = lane + 32 * k;
    float hn = (v[k] - mu) * inv * ln_g[layer * H_DIM + c] + ln_b[layer * H_DIM + c];
    hn = fmaxf(hn, 0.f);
    size_t oi = (size_t)n * H_DIM + c;
    float nh = (layer == 0) ? hn : h[oi] + hn;
    h[oi] = nh;
    h_bf[oi] = f2bf(nh);
  }
}

// ---------------- final MLP: relu(h@w1+b1)@w2+b2, 16 nodes per block ----------------
__global__ __launch_bounds__(256) void mlp_kernel(
    const unsigned short* __restrict__ h_bf, const unsigned short* __restrict__ w1T,
    const float* __restrict__ b1, const unsigned short* __restrict__ w2T,
    const float* __restrict__ b2, float* __restrict__ out) {
  __shared__ __attribute__((aligned(16))) unsigned short y1[16 * H_DIM]; // 4KB bf16
  int wv = threadIdx.x >> 5, lane = threadIdx.x & 31;
  int nIdx = lane & 15, half = lane >> 4;
  int n0 = blockIdx.x * 16;
  int eCol = wv * 16 + nIdx;
  v8f c = {};
  const unsigned short* Arow = h_bf + ((size_t)n0 + nIdx) * H_DIM;
#pragma unroll
  for (int ks = 0; ks < 4; ++ks) {
    v16bf a = frag_from_bf16(Arow + ks * 32, half);
    v16bf b = frag_from_bf16(w1T + (size_t)eCol * H_DIM + ks * 32, half);
    c = wmma_bf16(a, b, c);
  }
  float bias = b1[eCol];
#pragma unroll
  for (int r = 0; r < 8; ++r)
    y1[(r + 8 * half) * H_DIM + eCol] = f2bf(fmaxf(c[r] + bias, 0.f));
  __syncthreads();
  if (wv == 0) {
    v8f c2 = {};
#pragma unroll
    for (int ks = 0; ks < 4; ++ks) {
      v16bf a = frag_from_bf16(y1 + nIdx * H_DIM + ks * 32, half);
      v16bf b = frag_from_bf16(w2T + nIdx * H_DIM + ks * 32, half);
      c2 = wmma_bf16(a, b, c2);
    }
    if (nIdx < OUT_DIM) {
#pragma unroll
      for (int r = 0; r < 8; ++r) {
        int n = n0 + r + 8 * half;
        out[(size_t)n * OUT_DIM + nIdx] = c2[r] + b2[nIdx];
      }
    }
  }
}

// ---------------- launch ----------------
extern "C" void kernel_launch(void* const* d_in, const int* in_sizes, int n_in,
                              void* d_out, int out_size, void* d_ws, size_t ws_size,
                              hipStream_t stream) {
  const float*          x           = (const float*)d_in[0];
  const long long*      node_type   = (const long long*)d_in[3];
  const int*            group_src   = (const int*)d_in[4];
  const int*            group_dst   = (const int*)d_in[5];
  const unsigned char*  group_valid = (const unsigned char*)d_in[6];
  const float* inp_w = (const float*)d_in[7];
  const float* inp_b = (const float*)d_in[8];
  const float* in_w  = (const float*)d_in[9];
  const float* in_b  = (const float*)d_in[10];
  const float* out_w = (const float*)d_in[11];
  const float* out_b = (const float*)d_in[12];
  const float* ln_g  = (const float*)d_in[13];
  const float* ln_b  = (const float*)d_in[14];
  const float* w1    = (const float*)d_in[15];
  const float* b1    = (const float*)d_in[16];
  const float* w2    = (const float*)d_in[17];
  const float* b2    = (const float*)d_in[18];
  float* out = (float*)d_out;

  // workspace carve-out (256B aligned)
  auto align256 = [](size_t v) { return (v + 255) & ~(size_t)255; };
  char* ws = (char*)d_ws;
  size_t off = 0;
  const size_t hBytes   = (size_t)N_NODES * H_DIM * sizeof(float);
  const size_t hBfBytes = (size_t)N_NODES * H_DIM * sizeof(unsigned short);
  const size_t gBytes   = (size_t)NGRP * LMAXV * H_DIM * sizeof(unsigned short);
  float*          h       = (float*)(ws + off);          off += align256(hBytes);
  float*          h_acc   = (float*)(ws + off);          off += align256(hBytes);
  unsigned short* h_bf    = (unsigned short*)(ws + off); off += align256(hBfBytes);
  unsigned short* q_bf    = (unsigned short*)(ws + off); off += align256(gBytes);
  unsigned short* k_bf    = (unsigned short*)(ws + off); off += align256(gBytes);
  unsigned short* vT_bf   = (unsigned short*)(ws + off); off += align256(gBytes + 256); // pad: tile overread
  unsigned short* o_bf    = (unsigned short*)(ws + off); off += align256(gBytes);
  unsigned short* in_w_bf = (unsigned short*)(ws + off); off += align256((size_t)INW_CNT * 2);
  unsigned short* out_w_bf= (unsigned short*)(ws + off); off += align256((size_t)OUTW_CNT * 2);
  unsigned short* w1T     = (unsigned short*)(ws + off); off += align256((size_t)H_DIM * H_DIM * 2);
  unsigned short* w2T     = (unsigned short*)(ws + off); off += align256((size_t)16 * H_DIM * 2);
  (void)ws_size; (void)in_sizes; (void)n_in; (void)out_size;

  prep_weights<<<512, 256, 0, stream>>>(in_w, out_w, w1, w2, in_w_bf, out_w_bf, w1T, w2T);
  input_proj<<<(N_NODES + 7) / 8, 256, 0, stream>>>(x, node_type, inp_w, inp_b, h, h_bf);

  for (int li = 0; li < NLAYERS; ++li) {
    hipMemsetAsync(h_acc, 0, hBytes, stream);
    qkv_kernel<<<NGRP * LTILES, 256, 0, stream>>>(h_bf, group_src, group_dst, in_w_bf, in_b, li,
                                                  q_bf, k_bf, vT_bf);
    attn_kernel<<<NGRP * MTILES, 256, 0, stream>>>(q_bf, k_bf, vT_bf, group_valid, o_bf);
    outproj_kernel<<<NGRP * LTILES, 256, 0, stream>>>(o_bf, group_dst, group_valid,
                                                      out_w_bf, out_b, li, h_acc);
    ln_update<<<(N_NODES + 7) / 8, 256, 0, stream>>>(h_acc, ln_g, ln_b, li, h, h_bf);
  }
  mlp_kernel<<<N_NODES / 16, 256, 0, stream>>>(h_bf, w1T, b1, w2T, b2, out);
}
